// SelfAttention_26938034881046
// MI455X (gfx1250) — compile-verified, matrix operand order checked by reading, not run on hardware
//
#include <hip/hip_runtime.h>
#include <math.h>

typedef unsigned short u16;
typedef __attribute__((ext_vector_type(8)))  __bf16 v8bf;
typedef __attribute__((ext_vector_type(16))) __bf16 v16bf;
typedef __attribute__((ext_vector_type(8)))  float  v8f;

#define B_  2
#define S_  2048
#define D_  1024
#define H_  16
#define DH_ 64
#define KDIM 1024

// ---- optional CDNA5 async global->LDS path (probe via __has_builtin) ----
#if defined(__has_builtin)
#if __has_builtin(__builtin_amdgcn_global_load_async_to_lds_b128)
#define ASYNC_LDS 1
#endif
#if __has_builtin(__builtin_amdgcn_s_wait_asynccnt)
#define HAVE_WAIT_ASYNC 1
#endif
#endif

// builtin signature (from hipcc diagnostic): param0 = int4(vector_size 16) AS1*,
// LDS side = AS3. Use matching GCC-vector typedefs; typed pointers implicitly
// convert to void* in the same AS if the builtin is declared that way instead.
typedef int v4i __attribute__((vector_size(16)));
typedef v4i __attribute__((address_space(1))) v4i_g;
typedef v4i __attribute__((address_space(3))) v4i_l;

// copy 16B global -> LDS. On CDNA5 this lowers to global_load_async_to_lds_b128
// (ASYNCcnt-tracked, no VGPR round trip); otherwise plain load + ds_store.
// Note: low 32 bits of a generic LDS address are the LDS offset (ISA aperture rule),
// so the integer cast to an AS(3) pointer is exact.
__device__ __forceinline__ void cp16(u16* dst_lds, const u16* src_glob) {
#if defined(ASYNC_LDS)
  __builtin_amdgcn_global_load_async_to_lds_b128(
      (v4i_g*)(unsigned long long)src_glob,
      (v4i_l*)(unsigned int)(unsigned long long)dst_lds, 0, 0);
#else
  *reinterpret_cast<uint4*>(dst_lds) = *reinterpret_cast<const uint4*>(src_glob);
#endif
}

__device__ __forceinline__ void async_drain() {
#if defined(ASYNC_LDS) && defined(HAVE_WAIT_ASYNC)
  __builtin_amdgcn_s_wait_asynccnt(0);
#endif
}

__device__ __forceinline__ u16 f2bf(float f) {
  unsigned int u = __float_as_uint(f);
  unsigned int lsb = (u >> 16) & 1u;
  u += 0x7FFFu + lsb;
  return (u16)(u >> 16);
}

__device__ __forceinline__ v16bf load16(const u16* p0, const u16* p1) {
  v8bf lo = *reinterpret_cast<const v8bf*>(p0);
  v8bf hi = *reinterpret_cast<const v8bf*>(p1);
  return __builtin_shufflevector(lo, hi, 0,1,2,3,4,5,6,7,8,9,10,11,12,13,14,15);
}

__device__ __forceinline__ v8f vzero8() {
  v8f z = {0.f,0.f,0.f,0.f,0.f,0.f,0.f,0.f};
  return z;
}

// fp32 -> bf16, optional transpose (dst laid out [cols][rows], coalesced writes)
__global__ void convert_kernel(const float* __restrict__ src, u16* __restrict__ dst,
                               int rows, int cols, int do_transpose, int total) {
  int idx = blockIdx.x * blockDim.x + threadIdx.x;
  if (idx >= total) return;
  float v;
  if (do_transpose) {
    int r = idx % rows;       // k index
    int n = idx / rows;       // n index
    v = src[(size_t)r * cols + n];
  } else {
    v = src[idx];
  }
  dst[idx] = f2bf(v);
}

// C[M,N] = A[M,KDIM](bf16,row-major) * Bt[N,KDIM]^T (bf16) + bias
// mode 0: scatter bf16 into Q/K/V per-head [B,H,S,dh] layouts
// mode 1: write fp32 to Cout[M,N]
// Double-buffered LDS; tile staging via async global->LDS when available.
__global__ __launch_bounds__(256)
void gemm_bf16_wmma(const u16* __restrict__ A, const u16* __restrict__ Bt,
                    const float* __restrict__ bias,
                    int M, int N, int mode,
                    u16* __restrict__ Qb, u16* __restrict__ Kb, u16* __restrict__ Vb,
                    float* __restrict__ Cout) {
  __shared__ u16 sA[2][128 * 32];
  __shared__ u16 sB[2][128 * 32];
  const int tid  = threadIdx.x;
  const int wave = tid >> 5, lane = tid & 31;
  const int wm = wave >> 2, wn = wave & 3;     // 2x4 wave grid, wave tile 64x32
  const int rl = lane & 15, half = lane >> 4;
  const int koff = half ? 8 : 0;               // 16-bit A/B lane K-layout
  const int blockM = blockIdx.y * 128, blockN = blockIdx.x * 128;

  // per-thread copy slots (2 x 16B per tile per thread)
  const int crow0 = tid >> 2,            ccu0 = (tid & 3) << 3;
  const int crow1 = (tid + 256) >> 2,    ccu1 = ccu0;

  v8f acc[4][2];
  #pragma unroll
  for (int mt = 0; mt < 4; ++mt)
    #pragma unroll
    for (int nt = 0; nt < 2; ++nt)
      acc[mt][nt] = vzero8();

  // prologue: stage tile 0
  cp16(&sA[0][crow0 * 32 + ccu0], &A[(size_t)(blockM + crow0) * KDIM + ccu0]);
  cp16(&sB[0][crow0 * 32 + ccu0], &Bt[(size_t)(blockN + crow0) * KDIM + ccu0]);
  cp16(&sA[0][crow1 * 32 + ccu1], &A[(size_t)(blockM + crow1) * KDIM + ccu1]);
  cp16(&sB[0][crow1 * 32 + ccu1], &Bt[(size_t)(blockN + crow1) * KDIM + ccu1]);
  async_drain();
  __syncthreads();

  const int nk = KDIM / 32;
  for (int t = 0; t < nk; ++t) {
    const int cur = t & 1;
    // stage tile t+1 into the other buffer while computing on this one
    if (t + 1 < nk) {
      const int kc = (t + 1) * 32;
      const int nxt = cur ^ 1;
      cp16(&sA[nxt][crow0 * 32 + ccu0], &A[(size_t)(blockM + crow0) * KDIM + kc + ccu0]);
      cp16(&sB[nxt][crow0 * 32 + ccu0], &Bt[(size_t)(blockN + crow0) * KDIM + kc + ccu0]);
      cp16(&sA[nxt][crow1 * 32 + ccu1], &A[(size_t)(blockM + crow1) * KDIM + kc + ccu1]);
      cp16(&sB[nxt][crow1 * 32 + ccu1], &Bt[(size_t)(blockN + crow1) * KDIM + kc + ccu1]);
    }

    v16bf af[4], bfv[2];
    #pragma unroll
    for (int mt = 0; mt < 4; ++mt) {
      const u16* p = &sA[cur][(wm * 64 + mt * 16 + rl) * 32 + koff];
      af[mt] = load16(p, p + 16);
    }
    #pragma unroll
    for (int nt = 0; nt < 2; ++nt) {
      const u16* p = &sB[cur][(wn * 32 + nt * 16 + rl) * 32 + koff];
      bfv[nt] = load16(p, p + 16);
    }
    #pragma unroll
    for (int mt = 0; mt < 4; ++mt)
      #pragma unroll
      for (int nt = 0; nt < 2; ++nt)
        acc[mt][nt] = __builtin_amdgcn_wmma_f32_16x16x32_bf16(
            false, af[mt], false, bfv[nt], (short)0, acc[mt][nt], false, false);

    async_drain();       // only residual latency not hidden by the 8 WMMAs
    __syncthreads();
  }

  // epilogue: D layout: VGPR i -> row (i + 8*half), col = lane%16 (per 16x16 tile)
  #pragma unroll
  for (int mt = 0; mt < 4; ++mt)
    #pragma unroll
    for (int nt = 0; nt < 2; ++nt)
      #pragma unroll
      for (int i = 0; i < 8; ++i) {
        int m = blockM + wm * 64 + mt * 16 + i + half * 8;
        int c = blockN + wn * 32 + nt * 16 + rl;
        float v = acc[mt][nt][i] + bias[c];
        if (mode == 0) {
          int b = m >> 11, s = m & (S_ - 1);
          int which = c >> 10, d = c & (D_ - 1);
          int h = d >> 6, dd = d & (DH_ - 1);
          size_t dst = (((size_t)(b * H_ + h)) * S_ + s) * DH_ + dd;
          u16* outp = (which == 0) ? Qb : ((which == 1) ? Kb : Vb);
          outp[dst] = f2bf(v);
        } else {
          Cout[(size_t)m * N + c] = v;
        }
      }
}

// flash attention: grid (S/128, B*H); each wave owns 16 q rows
__global__ __launch_bounds__(256)
void attn_wmma(const u16* __restrict__ Qb, const u16* __restrict__ Kb,
               const u16* __restrict__ Vb, u16* __restrict__ A2) {
  __shared__ u16 sK[32 * 64];        // K chunk, row-major [k][d]
  __shared__ u16 sV[64 * 32];        // V chunk transposed [d][k]
  __shared__ u16 sP[8][16 * 32];     // per-wave P tile (D-layout -> A-layout hop)

  const int tid  = threadIdx.x;
  const int wave = tid >> 5, lane = tid & 31;
  const int rl = lane & 15, half = lane >> 4;
  const int koff = half ? 8 : 0;
  const int bh = blockIdx.y;
  const int b = bh / H_, h = bh % H_;
  const int q0 = blockIdx.x * 128 + wave * 16;
  const size_t base = (size_t)bh * S_ * DH_;
  const float scale = 0.125f;        // 1/sqrt(64)

  // Q fragments (A-layout), dh=64 -> 2 k-depth-32 fragments
  v16bf aq[2];
  #pragma unroll
  for (int kk = 0; kk < 2; ++kk) {
    const u16* p = &Qb[base + (size_t)(q0 + rl) * DH_ + kk * 32 + koff];
    aq[kk] = load16(p, p + 16);
  }

  v8f o[4];
  float mi[8], li[8];
  #pragma unroll
  for (int nt = 0; nt < 4; ++nt) o[nt] = vzero8();
  #pragma unroll
  for (int i = 0; i < 8; ++i) { mi[i] = -INFINITY; li[i] = 0.f; }

  const int nch = blockIdx.x * 4 + 4;   // causal: chunks up to block diagonal
  for (int c = 0; c < nch; ++c) {
    // cooperative load: K via async copy, V transposed through VGPRs
    {
      int krow = tid >> 3;
      int cu   = (tid & 7) << 3;
      cp16(&sK[krow * 64 + cu], &Kb[base + (size_t)(c * 32 + krow) * DH_ + cu]);
      uint4 vv = *reinterpret_cast<const uint4*>(&Vb[base + (size_t)(c * 32 + krow) * DH_ + cu]);
      const u16* ve = reinterpret_cast<const u16*>(&vv);
      #pragma unroll
      for (int j = 0; j < 8; ++j) sV[(cu + j) * 32 + krow] = ve[j];
    }
    if (c + 1 < nch) {
      __builtin_prefetch(&Kb[base + (size_t)((c + 1) * 32 + (tid & 31)) * DH_], 0, 1);
      __builtin_prefetch(&Vb[base + (size_t)((c + 1) * 32 + (tid & 31)) * DH_], 0, 1);
    }
    async_drain();
    __syncthreads();

    if (c * 32 <= q0 + 15) {           // wave-uniform: skip fully-masked chunks
      // S = Q @ K^T : 16x32 scores = 2 n-tiles x 2 k-depth WMMAs
      v8f sf[2] = { vzero8(), vzero8() };
      #pragma unroll
      for (int nt = 0; nt < 2; ++nt)
        #pragma unroll
        for (int kk = 0; kk < 2; ++kk) {
          const u16* p = &sK[(nt * 16 + rl) * 64 + kk * 32 + koff];
          v16bf bfrag = load16(p, p + 16);
          sf[nt] = __builtin_amdgcn_wmma_f32_16x16x32_bf16(
              false, aq[kk], false, bfrag, (short)0, sf[nt], false, false);
        }

      // scale + causal mask
      float sc[2][8];
      #pragma unroll
      for (int nt = 0; nt < 2; ++nt)
        #pragma unroll
        for (int i = 0; i < 8; ++i) {
          int kcol = c * 32 + nt * 16 + rl;
          int q = q0 + i + half * 8;
          float v = sf[nt][i] * scale;
          sc[nt][i] = (kcol > q) ? -INFINITY : v;
        }

      // online softmax: row max / rescale (reduction stays within 16-lane half)
      float mnew[8], alpha[8];
      #pragma unroll
      for (int i = 0; i < 8; ++i) {
        float t = fmaxf(sc[0][i], sc[1][i]);
        t = fmaxf(t, __shfl_xor(t, 1));
        t = fmaxf(t, __shfl_xor(t, 2));
        t = fmaxf(t, __shfl_xor(t, 4));
        t = fmaxf(t, __shfl_xor(t, 8));
        mnew[i]  = fmaxf(mi[i], t);
        alpha[i] = __expf(mi[i] - mnew[i]);
        mi[i]    = mnew[i];
      }
      float rs[8];
      #pragma unroll
      for (int i = 0; i < 8; ++i) rs[i] = 0.f;
      #pragma unroll
      for (int nt = 0; nt < 2; ++nt)
        #pragma unroll
        for (int i = 0; i < 8; ++i) {
          float p = __expf(sc[nt][i] - mnew[i]);
          sc[nt][i] = p;
          rs[i] += p;
        }
      #pragma unroll
      for (int i = 0; i < 8; ++i) {
        float r = rs[i];
        r += __shfl_xor(r, 1);
        r += __shfl_xor(r, 2);
        r += __shfl_xor(r, 4);
        r += __shfl_xor(r, 8);
        li[i] = li[i] * alpha[i] + r;
      }
      #pragma unroll
      for (int nt = 0; nt < 4; ++nt)
        #pragma unroll
        for (int i = 0; i < 8; ++i) o[nt][i] *= alpha[i];

      // P: D-layout -> A-layout via wave-local LDS (DS ops in-order per wave)
      u16* pp = sP[wave];
      #pragma unroll
      for (int nt = 0; nt < 2; ++nt)
        #pragma unroll
        for (int i = 0; i < 8; ++i)
          pp[(i + half * 8) * 32 + nt * 16 + rl] = f2bf(sc[nt][i]);
      const u16* pl = &pp[rl * 32 + koff];
      v16bf pf = load16(pl, pl + 16);

      // O += P @ V : dh=64 -> 4 n-tiles
      #pragma unroll
      for (int nt = 0; nt < 4; ++nt) {
        const u16* pv = &sV[(nt * 16 + rl) * 32 + koff];
        v16bf bfrag = load16(pv, pv + 16);
        o[nt] = __builtin_amdgcn_wmma_f32_16x16x32_bf16(
            false, pf, false, bfrag, (short)0, o[nt], false, false);
      }
    }
    __syncthreads();
  }

  // epilogue: O /= l, re-interleave heads into [B,S,D] bf16
  #pragma unroll
  for (int nt = 0; nt < 4; ++nt)
    #pragma unroll
    for (int i = 0; i < 8; ++i) {
      int q = q0 + i + half * 8;
      int d = nt * 16 + rl;
      A2[((size_t)b * S_ + q) * D_ + h * DH_ + d] = f2bf(o[nt][i] / li[i]);
    }
}

extern "C" void kernel_launch(void* const* d_in, const int* in_sizes, int n_in,
                              void* d_out, int out_size, void* d_ws, size_t ws_size,
                              hipStream_t stream) {
  const float* x     = (const float*)d_in[0];
  const float* w_in  = (const float*)d_in[1];
  const float* b_in  = (const float*)d_in[2];
  const float* w_out = (const float*)d_in[3];
  const float* b_out = (const float*)d_in[4];
  // d_in[5] = causal_mask scalar (reference fixes it to 1)

  char* ws = (char*)d_ws;
  const size_t MB = 1024 * 1024;
  u16* Xb  = (u16*)(ws + 0 * MB);   // [4096,1024]  bf16  (8 MB)
  u16* Wti = (u16*)(ws + 8 * MB);   // [3072,1024]  bf16  (6 MB)  w_in^T
  u16* Wto = (u16*)(ws + 14 * MB);  // [1024,1024]  bf16  (2 MB)  w_out^T
  u16* Qb  = (u16*)(ws + 16 * MB);  // [B,H,S,64]   bf16  (8 MB)
  u16* Kb  = (u16*)(ws + 24 * MB);  // [B,H,S,64]   bf16  (8 MB)
  u16* Vb  = (u16*)(ws + 32 * MB);  // [B,H,S,64]   bf16  (8 MB)
  u16* A2  = (u16*)(ws + 40 * MB);  // [4096,1024]  bf16  (8 MB)
  float* out = (float*)d_out;

  int tx = B_ * S_ * D_;
  convert_kernel<<<(tx + 255) / 256, 256, 0, stream>>>(x, Xb, D_, 3 * D_, 0, tx);
  int tw = D_ * 3 * D_;
  convert_kernel<<<(tw + 255) / 256, 256, 0, stream>>>(w_in, Wti, D_, 3 * D_, 1, tw);
  int tw2 = D_ * D_;
  convert_kernel<<<(tw2 + 255) / 256, 256, 0, stream>>>(w_out, Wto, D_, D_, 1, tw2);

  dim3 g1(3 * D_ / 128, B_ * S_ / 128);
  gemm_bf16_wmma<<<g1, 256, 0, stream>>>(Xb, Wti, b_in, B_ * S_, 3 * D_, 0,
                                         Qb, Kb, Vb, nullptr);
  dim3 g2(S_ / 128, B_ * H_);
  attn_wmma<<<g2, 256, 0, stream>>>(Qb, Kb, Vb, A2);

  dim3 g3(D_ / 128, B_ * S_ / 128);
  gemm_bf16_wmma<<<g3, 256, 0, stream>>>(A2, Wto, b_out, B_ * S_, D_, 1,
                                         nullptr, nullptr, nullptr, out);
}
